// MoEBlock_16819091931327
// MI455X (gfx1250) — compile-verified, matrix operand order checked by reading, not run on hardware
//
#include <hip/hip_runtime.h>
#include <hip/hip_bf16.h>
#include <stdint.h>

typedef __attribute__((ext_vector_type(16))) __bf16 v16bf;
typedef __attribute__((ext_vector_type(8)))  float  v8f;
typedef __attribute__((ext_vector_type(4)))  unsigned int v4u;
typedef __attribute__((ext_vector_type(8)))  int  v8i;
typedef __attribute__((ext_vector_type(4)))  int  v4i;

// Fixed problem dims from the reference setup_inputs()
constexpr int T = 4 * 2048;      // B*S = 8192 tokens
constexpr int D = 2048;
constexpr int H = 4096;
constexpr int E = 8;
constexpr int TILE_M = 16;
constexpr int TILES_MAX = T / TILE_M;  // 512

#if __has_builtin(__builtin_amdgcn_tensor_load_to_lds)
#define MOE_USE_TDM 1
#else
#define MOE_USE_TDM 0
#endif

__device__ __forceinline__ float silu_fast(float v) {
  return v * __builtin_amdgcn_rcpf(1.0f + __expf(-v));  // x * v_rcp(1+e^-x)
}

// ---------------------------------------------------------------------------
__global__ void zero_f32(float* __restrict__ p, size_t n) {
  size_t i = (size_t)blockIdx.x * blockDim.x + threadIdx.x;
  size_t stride = (size_t)gridDim.x * blockDim.x;
  for (; i < n; i += stride) p[i] = 0.0f;
}

// ---------------------------------------------------------------------------
// Router: one wave32 per token.
__global__ void router_kernel(const float* __restrict__ x,
                              const float* __restrict__ Wg,
                              int* __restrict__ topi, float* __restrict__ topg) {
  const int wave = threadIdx.x >> 5;
  const int lane = threadIdx.x & 31;
  const int t = blockIdx.x * 8 + wave;
  if (t >= T) return;

  float acc[E];
#pragma unroll
  for (int e = 0; e < E; ++e) acc[e] = 0.0f;

  const float* xr = x + (size_t)t * D;
  for (int d = lane; d < D; d += 32) {
    float xv = xr[d];
    const float* wr = Wg + (size_t)d * E;
#pragma unroll
    for (int e = 0; e < E; ++e) acc[e] += xv * wr[e];
  }
#pragma unroll
  for (int e = 0; e < E; ++e)
    for (int off = 16; off > 0; off >>= 1)
      acc[e] += __shfl_xor(acc[e], off, 32);

  if (lane == 0) {
    float m = acc[0];
#pragma unroll
    for (int e = 1; e < E; ++e) m = fmaxf(m, acc[e]);
    float p[E];
#pragma unroll
    for (int e = 0; e < E; ++e) p[e] = __expf(acc[e] - m);
    int i0 = 0;
#pragma unroll
    for (int e = 1; e < E; ++e) if (p[e] > p[i0]) i0 = e;
    int i1 = (i0 == 0) ? 1 : 0;
#pragma unroll
    for (int e = 0; e < E; ++e) if (e != i0 && p[e] > p[i1]) i1 = e;
    float gs = p[i0] + p[i1];
    topi[2 * t + 0] = i0;            topi[2 * t + 1] = i1;
    topg[2 * t + 0] = p[i0] / gs;    topg[2 * t + 1] = p[i1] / gs;
  }
}

// ---------------------------------------------------------------------------
// Deterministic per-expert token compaction (block-wide scan, no atomics).
__global__ void build_lists_kernel(const int* __restrict__ topi,
                                   const float* __restrict__ topg,
                                   int* __restrict__ counts,
                                   int* __restrict__ lists,
                                   float* __restrict__ lgates) {
  const int e = blockIdx.x;
  const int tid = threadIdx.x;
  __shared__ int sc[256];
  int base = 0;
  for (int t0 = 0; t0 < T; t0 += 256) {
    int t = t0 + tid;
    int f = 0; float g = 0.0f;
    int i0 = topi[2 * t], i1 = topi[2 * t + 1];
    if (i0 == e)      { f = 1; g = topg[2 * t]; }
    else if (i1 == e) { f = 1; g = topg[2 * t + 1]; }
    sc[tid] = f;
    __syncthreads();
    for (int off = 1; off < 256; off <<= 1) {
      int v = (tid >= off) ? sc[tid - off] : 0;
      __syncthreads();
      sc[tid] += v;
      __syncthreads();
    }
    int pos = base + sc[tid] - f;
    int total = sc[255];
    if (f) { lists[e * T + pos] = t; lgates[e * T + pos] = g; }
    base += total;
    __syncthreads();
  }
  if (tid == 0) counts[e] = base;
}

// ---------------------------------------------------------------------------
// Pre-convert fp32 weights into bf16 arranged as WMMA B fragments:
//   dst[e][nt][kt][lane][j]  (j = 0..15, the lane's 16 fragment elements)
__global__ void swizzle_weights_kernel(const float* __restrict__ W,
                                       __bf16* __restrict__ Wb,
                                       int Kdim, int Ndim) {
  const int KT = Kdim / 32, NT = Ndim / 16;
  size_t gid = (size_t)blockIdx.x * blockDim.x + threadIdx.x;
  size_t total = (size_t)E * KT * NT * 32;
  if (gid >= total) return;
  int lane = (int)(gid & 31);
  size_t fb = gid >> 5;
  int kt = (int)(fb % KT);
  size_t r1 = fb / KT;
  int nt = (int)(r1 % NT);
  int e  = (int)(r1 / NT);

  const int n  = lane & 15;
  const int kh = (lane >> 4) << 3;
  const float* src = W + ((size_t)e * Kdim + (size_t)kt * 32 + kh) * Ndim + nt * 16 + n;
  __bf16* dst = Wb + ((((size_t)e * NT + nt) * KT + kt) * 32 + lane) * 16;
#pragma unroll
  for (int j = 0; j < 8; ++j) dst[j]     = (__bf16)src[(size_t)j * Ndim];
#pragma unroll
  for (int j = 0; j < 8; ++j) dst[8 + j] = (__bf16)src[(size_t)(16 + j) * Ndim];
}

// ---------------------------------------------------------------------------
__device__ __forceinline__ v16bf frag_a_lds(const __bf16* __restrict__ base,
                                            int lda, int kbase, int lane) {
  const __bf16* r = base + (lane & 15) * lda + kbase + ((lane >> 4) << 3);
  union { uint4 u[2]; v16bf v; } t;
  t.u[0] = *(const uint4*)(r);
  t.u[1] = *(const uint4*)(r + 16);
  return t.v;
}

__device__ __forceinline__ v16bf frag_b_swz(const __bf16* __restrict__ p) {
  union { uint4 u[2]; v16bf v; } t;
  const uint4* q = (const uint4*)p;
  t.u[0] = q[0];
  t.u[1] = q[1];
  return t.v;
}

__device__ __forceinline__ v16bf frag_b_global(const float* __restrict__ W,
                                               int ldb, int kbase, int nbase, int lane) {
  const int n = lane & 15;
  const int kh = (lane >> 4) << 3;
  const float* c = W + (size_t)(kbase + kh) * ldb + nbase + n;
  v16bf b;
#pragma unroll
  for (int v = 0; v < 8; ++v) b[v]     = (__bf16)c[(size_t)v * ldb];
#pragma unroll
  for (int v = 0; v < 8; ++v) b[8 + v] = (__bf16)c[(size_t)(16 + v) * ldb];
  return b;
}

#define WMMA_BF16(a, b, c) \
  __builtin_amdgcn_wmma_f32_16x16x32_bf16(false, (a), false, (b), (short)0, (c), false, false)

// ---------------------------------------------------------------------------
// Main expert kernel. Block = 16 routed tokens of one expert, 8 waves.
// Token rows gathered by one TDM gather-mode DMA (or per-lane async copies),
// both FFN layers fused with silu(h) resident in the 320KB WGP LDS.
// Each wave register-blocks 4 N-tiles: 1 A-fragment feeds 4 WMMAs.
__global__ void moe_expert_swz_kernel(const float* __restrict__ x,
                                      const __bf16* __restrict__ W1b,
                                      const __bf16* __restrict__ W2b,
                                      const int* __restrict__ counts,
                                      const int* __restrict__ lists,
                                      const float* __restrict__ lgates,
                                      float* __restrict__ out) {
  const int e    = blockIdx.x / TILES_MAX;
  const int tile = blockIdx.x % TILES_MAX;
  const int count = counts[e];
  if (tile * TILE_M >= count) return;

  extern __shared__ __bf16 smem[];
  __bf16* xs = smem;                    // [16][D] bf16   (64 KB)
  __bf16* hs = smem + TILE_M * D;       // [16][H] bf16   (128 KB)
  float*  xstage = (float*)hs;          // [16][D] fp32 staging, aliases hs
  __shared__ int   toks[TILE_M];
  __shared__ float gts[TILE_M];

  const int tid  = threadIdx.x;
  const int lane = tid & 31;
  const int wave = tid >> 5;

  if (tid < TILE_M) {
    int idx = tile * TILE_M + tid;
    if (idx < count) { toks[tid] = lists[e * T + idx]; gts[tid] = lgates[e * T + idx]; }
    else             { toks[tid] = -1;                 gts[tid] = 0.0f; }
  }
  __syncthreads();

  // --- gather 16 token rows (fp32) into LDS staging -------------------------
#if MOE_USE_TDM
  if (wave == 0) {
    // Tensor DMA descriptor, gather mode, 16-bit row indices (ISA 8.3-8.7).
    const uint64_t ga = (uint64_t)(uintptr_t)x;
    v4u g0;
    g0[0] = 0x80000001u;                               // count=1, gather_mode=1, 16b idx
    g0[1] = (unsigned)(uintptr_t)xstage;               // lds_addr
    g0[2] = (unsigned)(ga & 0xFFFFFFFFu);              // global_addr[31:0]
    g0[3] = ((unsigned)(ga >> 32) & 0x01FFFFFFu) | (2u << 30);  // addr[56:32], type=2
    v8i g1;
    g1[0] = 0x00020000;                                // data_size=2 (4B), mask=0
    g1[1] = (int)((unsigned)(D & 0xFFFF) << 16);       // tensor_dim0 lo<<16
    g1[2] = (int)(((unsigned)D >> 16) | ((unsigned)(T & 0xFFFF) << 16));  // dim0 hi | dim1 lo
    g1[3] = (int)(((unsigned)T >> 16) | ((unsigned)D << 16));  // dim1 hi | tile_dim0
    g1[4] = TILE_M;                                    // tile_dim1 = #indices (16)
    g1[5] = D;                                         // tensor_dim0_stride lo
    g1[6] = 0;
    g1[7] = 0;
    v4i g2, g3;
#pragma unroll
    for (int j = 0; j < 4; ++j) {
      int a0 = toks[2 * j] < 0 ? 0 : toks[2 * j];
      int a1 = toks[2 * j + 1] < 0 ? 0 : toks[2 * j + 1];
      g2[j] = (a0 & 0xFFFF) | (a1 << 16);
      int b0 = toks[8 + 2 * j] < 0 ? 0 : toks[8 + 2 * j];
      int b1 = toks[9 + 2 * j] < 0 ? 0 : toks[9 + 2 * j];
      g3[j] = (b0 & 0xFFFF) | (b1 << 16);
    }
#if __clang_major__ >= 23
    __builtin_amdgcn_tensor_load_to_lds(g0, g1, g2, g3, (v8i)0, 0);
#else
    __builtin_amdgcn_tensor_load_to_lds(g0, g1, g2, g3, 0);
#endif
    __builtin_amdgcn_s_wait_tensorcnt(0);
  }
  __syncthreads();
#else
  {
    const uint32_t lds_base = (uint32_t)(uintptr_t)xstage;
    for (int seg = tid; seg < TILE_M * D / 4; seg += 256) {  // 16B segments
      int r  = seg >> 9;
      int c4 = seg & 511;
      int tok = toks[r];
      const float* src = (tok >= 0) ? (x + (size_t)tok * D + (size_t)c4 * 4) : x;
      uint32_t dst = lds_base + (uint32_t)seg * 16u;
      uint64_t gaddr = (uint64_t)(uintptr_t)src;
      asm volatile("global_load_async_to_lds_b128 %0, %1, off"
                   :: "v"(dst), "v"(gaddr) : "memory");
    }
    asm volatile("s_wait_asynccnt 0x0" ::: "memory");
    __syncthreads();
  }
#endif
  // convert fp32 staging -> bf16 xs (zero-fill invalid rows)
  for (int i = tid; i < TILE_M * D; i += 256) {
    int r = i >> 11;
    float v = (toks[r] >= 0) ? xstage[i] : 0.0f;
    xs[i] = (__bf16)v;
  }
  __syncthreads();   // staging dead; hs region reusable

  const int KT1 = D / 32, NT1 = H / 16;
  const int KT2 = H / 32, NT2 = D / 16;
  const __bf16* W1e = W1b + (size_t)e * NT1 * KT1 * 512;
  const __bf16* W2e = W2b + (size_t)e * NT2 * KT2 * 512;

  // --- layer 1: h = silu(x @ W1), 4 N-tiles per wave ------------------------
  for (int nt0 = wave * 4; nt0 < NT1; nt0 += 32) {
    const __bf16* b0 = W1e + (size_t)nt0 * KT1 * 512 + lane * 16;
    const __bf16* b1 = b0 + (size_t)KT1 * 512;
    const __bf16* b2 = b1 + (size_t)KT1 * 512;
    const __bf16* b3 = b2 + (size_t)KT1 * 512;
    v8f acc0 = {}, acc1 = {}, acc2 = {}, acc3 = {};
    for (int kt = 0; kt < KT1; ++kt) {
      const size_t o = (size_t)kt * 512;
      __builtin_prefetch(b0 + o + 4096, 0, 1);
      __builtin_prefetch(b1 + o + 4096, 0, 1);
      __builtin_prefetch(b2 + o + 4096, 0, 1);
      __builtin_prefetch(b3 + o + 4096, 0, 1);
      v16bf a = frag_a_lds(xs, D, kt * 32, lane);
      acc0 = WMMA_BF16(a, frag_b_swz(b0 + o), acc0);
      acc1 = WMMA_BF16(a, frag_b_swz(b1 + o), acc1);
      acc2 = WMMA_BF16(a, frag_b_swz(b2 + o), acc2);
      acc3 = WMMA_BF16(a, frag_b_swz(b3 + o), acc3);
    }
    const int m0 = (lane >> 4) << 3;
    const int nc = lane & 15;
#pragma unroll
    for (int v = 0; v < 8; ++v) {
      const int m = v + m0;
      hs[m * H + (nt0 + 0) * 16 + nc] = (__bf16)silu_fast(acc0[v]);
      hs[m * H + (nt0 + 1) * 16 + nc] = (__bf16)silu_fast(acc1[v]);
      hs[m * H + (nt0 + 2) * 16 + nc] = (__bf16)silu_fast(acc2[v]);
      hs[m * H + (nt0 + 3) * 16 + nc] = (__bf16)silu_fast(acc3[v]);
    }
  }
  __syncthreads();

  // --- layer 2: out += gate * (h @ W2), 4 N-tiles per wave ------------------
  for (int nt0 = wave * 4; nt0 < NT2; nt0 += 32) {
    const __bf16* b0 = W2e + (size_t)nt0 * KT2 * 512 + lane * 16;
    const __bf16* b1 = b0 + (size_t)KT2 * 512;
    const __bf16* b2 = b1 + (size_t)KT2 * 512;
    const __bf16* b3 = b2 + (size_t)KT2 * 512;
    v8f acc0 = {}, acc1 = {}, acc2 = {}, acc3 = {};
    for (int kt = 0; kt < KT2; ++kt) {
      const size_t o = (size_t)kt * 512;
      __builtin_prefetch(b0 + o + 4096, 0, 1);
      __builtin_prefetch(b1 + o + 4096, 0, 1);
      __builtin_prefetch(b2 + o + 4096, 0, 1);
      __builtin_prefetch(b3 + o + 4096, 0, 1);
      v16bf a = frag_a_lds(hs, H, kt * 32, lane);
      acc0 = WMMA_BF16(a, frag_b_swz(b0 + o), acc0);
      acc1 = WMMA_BF16(a, frag_b_swz(b1 + o), acc1);
      acc2 = WMMA_BF16(a, frag_b_swz(b2 + o), acc2);
      acc3 = WMMA_BF16(a, frag_b_swz(b3 + o), acc3);
    }
    const int m0 = (lane >> 4) << 3;
    const int nc = lane & 15;
#pragma unroll
    for (int v = 0; v < 8; ++v) {
      const int m = v + m0;
      const int tok = toks[m];
      if (tok >= 0) {
        float* orow = out + (size_t)tok * D + nc;
        const float g = gts[m];
        atomicAdd(orow + (nt0 + 0) * 16, g * acc0[v]);
        atomicAdd(orow + (nt0 + 1) * 16, g * acc1[v]);
        atomicAdd(orow + (nt0 + 2) * 16, g * acc2[v]);
        atomicAdd(orow + (nt0 + 3) * 16, g * acc3[v]);
      }
    }
  }
}

// ---------------------------------------------------------------------------
// Fallback expert kernel (fp32 weights straight from global; ws too small).
__global__ void moe_expert_fb_kernel(const float* __restrict__ x,
                                     const float* __restrict__ W1,
                                     const float* __restrict__ W2,
                                     const int* __restrict__ counts,
                                     const int* __restrict__ lists,
                                     const float* __restrict__ lgates,
                                     float* __restrict__ out) {
  const int e    = blockIdx.x / TILES_MAX;
  const int tile = blockIdx.x % TILES_MAX;
  const int count = counts[e];
  if (tile * TILE_M >= count) return;

  extern __shared__ __bf16 smem[];
  __bf16* xs = smem;
  __bf16* hs = smem + TILE_M * D;
  __shared__ int   toks[TILE_M];
  __shared__ float gts[TILE_M];

  const int tid = threadIdx.x, lane = tid & 31, wave = tid >> 5;

  if (tid < TILE_M) {
    int idx = tile * TILE_M + tid;
    if (idx < count) { toks[tid] = lists[e * T + idx]; gts[tid] = lgates[e * T + idx]; }
    else             { toks[tid] = -1;                 gts[tid] = 0.0f; }
  }
  __syncthreads();
  for (int i = tid; i < TILE_M * D; i += 256) {
    int r = i >> 11, c = i & (D - 1);
    int tok = toks[r];
    xs[i] = (__bf16)((tok >= 0) ? x[(size_t)tok * D + c] : 0.0f);
  }
  __syncthreads();

  const float* W1e = W1 + (size_t)e * D * H;
  const float* W2e = W2 + (size_t)e * H * D;

  for (int nt = wave; nt < H / 16; nt += 8) {
    const int nbase = nt * 16;
    v8f acc = {};
    for (int k = 0; k < D; k += 32) {
      v16bf a = frag_a_lds(xs, D, k, lane);
      v16bf b = frag_b_global(W1e, H, k, nbase, lane);
      acc = WMMA_BF16(a, b, acc);
    }
    const int n = nbase + (lane & 15);
#pragma unroll
    for (int v = 0; v < 8; ++v) {
      const int m = v + ((lane >> 4) << 3);
      hs[m * H + n] = (__bf16)silu_fast(acc[v]);
    }
  }
  __syncthreads();

  for (int nt = wave; nt < D / 16; nt += 8) {
    const int nbase = nt * 16;
    v8f acc = {};
    for (int k = 0; k < H; k += 32) {
      v16bf a = frag_a_lds(hs, H, k, lane);
      v16bf b = frag_b_global(W2e, D, k, nbase, lane);
      acc = WMMA_BF16(a, b, acc);
    }
    const int n = nbase + (lane & 15);
#pragma unroll
    for (int v = 0; v < 8; ++v) {
      const int m = v + ((lane >> 4) << 3);
      const int tok = toks[m];
      if (tok >= 0) atomicAdd(&out[(size_t)tok * D + n], gts[m] * acc[v]);
    }
  }
}

// ---------------------------------------------------------------------------
extern "C" void kernel_launch(void* const* d_in, const int* in_sizes, int n_in,
                              void* d_out, int out_size, void* d_ws, size_t ws_size,
                              hipStream_t stream) {
  (void)in_sizes; (void)n_in; (void)out_size;
  const float* x  = (const float*)d_in[0];
  const float* Wg = (const float*)d_in[1];
  const float* W1 = (const float*)d_in[2];
  const float* W2 = (const float*)d_in[3];
  float* out = (float*)d_out;

  char* ws = (char*)d_ws;
  size_t off = 0;
  auto alloc = [&](size_t bytes) -> void* {
    void* p = ws + off;
    off = (off + bytes + 255) & ~(size_t)255;
    return p;
  };
  int*    topi   = (int*)   alloc(sizeof(int)   * (size_t)T * 2);
  float*  topg   = (float*) alloc(sizeof(float) * (size_t)T * 2);
  int*    counts = (int*)   alloc(sizeof(int)   * E);
  int*    lists  = (int*)   alloc(sizeof(int)   * (size_t)E * T);
  float*  lgates = (float*) alloc(sizeof(float) * (size_t)E * T);
  __bf16* W1b    = (__bf16*)alloc(sizeof(__bf16) * (size_t)E * D * H);
  __bf16* W2b    = (__bf16*)alloc(sizeof(__bf16) * (size_t)E * H * D);
  const bool have_bf = (off <= ws_size);   // fixed per harness -> deterministic

  zero_f32<<<1024, 256, 0, stream>>>(out, (size_t)T * D);
  router_kernel<<<T / 8, 256, 0, stream>>>(x, Wg, topi, topg);
  build_lists_kernel<<<E, 256, 0, stream>>>(topi, topg, counts, lists, lgates);

  const size_t lds_bytes = (size_t)TILE_M * (D + H) * sizeof(__bf16);  // 192 KB

  if (have_bf) {
    const size_t thr1 = (size_t)E * (D / 32) * (H / 16) * 32;
    const size_t thr2 = (size_t)E * (H / 32) * (D / 16) * 32;
    swizzle_weights_kernel<<<(int)((thr1 + 255) / 256), 256, 0, stream>>>(W1, W1b, D, H);
    swizzle_weights_kernel<<<(int)((thr2 + 255) / 256), 256, 0, stream>>>(W2, W2b, H, D);
    moe_expert_swz_kernel<<<E * TILES_MAX, 256, lds_bytes, stream>>>(
        x, W1b, W2b, counts, lists, lgates, out);
  } else {
    moe_expert_fb_kernel<<<E * TILES_MAX, 256, lds_bytes, stream>>>(
        x, W1, W2, counts, lists, lgates, out);
  }
}
